// NequIPESCN_9079560864096
// MI455X (gfx1250) — matrix operand order repaired and checked by reading, not compiled, not run on hardware
//
#include <hip/hip_runtime.h>
#include <hip/hip_bf16.h>

typedef __attribute__((ext_vector_type(16))) _Float16 v16h;
typedef __attribute__((ext_vector_type(8)))  _Float16 v8h;
typedef __attribute__((ext_vector_type(8)))  float    v8f;

#define N_NODES 32768
#define N_EDGES 524288
#define FEAT    144
#define PI_F    3.14159265358979f

// ---------------- Clebsch-Gordan sparse tables (real basis, l<=2) -----------
// l=1 component order: (y, z, x).  l=2 order: (xy, yz, z2, xz, x2-y2).
constexpr int PL1[11] = {0,1,2,0,1,1,2,0,1,2,2};
constexpr int PL2[11] = {0,1,2,1,0,2,1,2,1,0,2};
constexpr int PL3[11] = {0,0,0,1,1,1,1,2,2,2,2};
constexpr float PINVN[3] = {0.57735026919f, 0.5f, 0.5f}; // 1/PATH_NORM[l3]
constexpr int PSTART[12] = {0,1,4,9,12,15,26,37,42,53,58,83};

constexpr unsigned char CGI[83] = {
  /*P0*/0,
  /*P1*/0,1,2,
  /*P2*/0,1,2,3,4,
  /*P3*/0,0,0,
  /*P4*/0,1,2,
  /*P5*/2,0,0,1,1,2,0,2,1,2,0,
  /*P6*/0,0,1,1,2,2,2,3,3,4,4,
  /*P7*/0,0,0,0,0,
  /*P8*/2,0,0,1,1,2,0,2,1,2,0,
  /*P9*/0,1,2,3,4,
  /*P10*/0,0,1,1,3,3, 0,0,2, 1,1,2, 3,3,2, 1,1,4, 3,3,4, 2, 2,4,4};
constexpr unsigned char CGJ[83] = {
  /*P0*/0,
  /*P1*/0,1,2,
  /*P2*/0,1,2,3,4,
  /*P3*/0,1,2,
  /*P4*/0,0,0,
  /*P5*/0,0,1,1,2,2,2,3,3,4,4,
  /*P6*/2,0,0,1,1,2,0,2,1,2,0,
  /*P7*/0,1,2,3,4,
  /*P8*/0,2,1,0,1,2,0,1,2,2,0,
  /*P9*/0,0,0,0,0,
  /*P10*/1,3,0,3,0,1, 0,2,0, 1,2,1, 3,2,3, 1,4,1, 3,4,3, 2, 4,2,4};
constexpr unsigned char CGK[83] = {
  /*P0*/0,
  /*P1*/0,0,0,
  /*P2*/0,0,0,0,0,
  /*P3*/0,1,2,
  /*P4*/0,1,2,
  /*P5*/0,2,1,0,1,2,0,1,2,2,0,
  /*P6*/0,2,1,0,1,2,0,1,2,2,0,
  /*P7*/0,1,2,3,4,
  /*P8*/0,0,1,1,2,2,2,3,3,4,4,
  /*P9*/0,1,2,3,4,
  /*P10*/3,1,3,0,1,0, 2,0,0, 2,1,1, 2,3,3, 2,1,1, 2,3,3, 2, 4,4,2};
constexpr float CGV[83] = {
  /*P0*/1.0f,
  /*P1*/0.57735026919f,0.57735026919f,0.57735026919f,
  /*P2*/0.44721359550f,0.44721359550f,0.44721359550f,0.44721359550f,0.44721359550f,
  /*P3*/1.0f,1.0f,1.0f,
  /*P4*/1.0f,1.0f,1.0f,
  /*P5*/0.54772255751f,0.54772255751f,0.54772255751f,0.54772255751f,0.63245553203f,
       -0.31622776602f,-0.31622776602f,0.54772255751f,0.54772255751f,0.54772255751f,-0.54772255751f,
  /*P6*/0.54772255751f,0.54772255751f,0.54772255751f,0.54772255751f,0.63245553203f,
       -0.31622776602f,-0.31622776602f,0.54772255751f,0.54772255751f,0.54772255751f,-0.54772255751f,
  /*P7*/1.0f,1.0f,1.0f,1.0f,1.0f,
  /*P8*/0.70710678119f,0.70710678119f,0.70710678119f,0.70710678119f,0.81649658092f,
       -0.40824829046f,-0.40824829046f,0.70710678119f,0.70710678119f,0.70710678119f,-0.70710678119f,
  /*P9*/1.0f,1.0f,1.0f,1.0f,1.0f,
  /*P10*/0.46291004988f,0.46291004988f,0.46291004988f,0.46291004988f,0.46291004988f,0.46291004988f,
        -0.53452248382f,-0.53452248382f,-0.53452248382f,
         0.26726124191f,0.26726124191f,0.26726124191f,
         0.26726124191f,0.26726124191f,0.26726124191f,
        -0.46291004988f,-0.46291004988f,-0.46291004988f,
         0.46291004988f,0.46291004988f,0.46291004988f,
         0.53452248382f,
        -0.53452248382f,-0.53452248382f,-0.53452248382f};

// fast silu: v_exp_f32 + v_add + v_rcp_f32 + v_mul (avoids IEEE div chain)
__device__ __forceinline__ float silu_f(float x) {
  return x * __builtin_amdgcn_rcpf(1.0f + __expf(-x));
}

// ---------------- weight pre-swizzle into WMMA B-fragment layout ------------
// B frag (16x16x32 f16): lane l holds col = l&15, K = (e>>3)*16 + (l>>4)*8 + (e&7), e=0..15.
// frags 0..3 : Wm0 (K=8 padded to 32, 4 col tiles), scale 1/sqrt(8)
// frags 4..11: Wm1 (t=0..3, c=0..1), scale 1/8
// frags 12..33: Wm2 (p=0..10, c=0..1), scale 1/8
__global__ void nequip_frag_kernel(const float* __restrict__ Wm0,
                                   const float* __restrict__ Wm1,
                                   const float* __restrict__ Wm2,
                                   _Float16* __restrict__ bf) {
  int idx = blockIdx.x * blockDim.x + threadIdx.x;
  if (idx >= 34 * 512) return;
  int frag = idx >> 9;
  int rem  = idx & 511;
  int lane = rem >> 4;
  int e    = rem & 15;
  const float* W; int Kdim, Ndim, t, c; float scale;
  if (frag < 4)       { t = frag;        c = 0;          W = Wm0; Kdim = 8;  Ndim = 64;  scale = 0.35355339059f; }
  else if (frag < 12) { int f = frag-4;  t = f>>1; c = f&1; W = Wm1; Kdim = 64; Ndim = 64;  scale = 0.125f; }
  else                { int f = frag-12; t = f>>1; c = f&1; W = Wm2; Kdim = 64; Ndim = 176; scale = 0.125f; }
  int k   = c*32 + ((e>>3)<<4) + ((lane>>4)<<3) + (e&7);
  int col = t*16 + (lane & 15);
  float v = (k < Kdim) ? W[k*Ndim + col] * scale : 0.0f;
  bf[idx] = (_Float16)v;
}

// ---------------- node up-projection: m = f @ W_up * 1/sqrt(CH) -------------
__global__ void nequip_up_kernel(const float* __restrict__ nf,
                                 const float* __restrict__ Wu0,
                                 const float* __restrict__ Wu1,
                                 const float* __restrict__ Wu2,
                                 float* __restrict__ m) {
  int idx = blockIdx.x * blockDim.x + threadIdx.x;
  if (idx >= N_NODES * 16) return;
  int n = idx >> 4, o = idx & 15;
  const float* f = nf + (size_t)n * FEAT;
  float* mo = m + (size_t)n * FEAT;
  const float inv = 0.25f;
  float s0 = 0.f;
  #pragma unroll
  for (int c = 0; c < 16; ++c) s0 += f[c] * Wu0[c*16 + o];
  mo[o] = s0 * inv;
  #pragma unroll
  for (int i = 0; i < 3; ++i) {
    float s = 0.f;
    #pragma unroll
    for (int c = 0; c < 16; ++c) s += f[16 + c*3 + i] * Wu1[c*16 + o];
    mo[16 + o*3 + i] = s * inv;
  }
  #pragma unroll
  for (int i = 0; i < 5; ++i) {
    float s = 0.f;
    #pragma unroll
    for (int c = 0; c < 16; ++c) s += f[64 + c*5 + i] * Wu2[c*16 + o];
    mo[64 + o*5 + i] = s * inv;
  }
}

__global__ void nequip_zero_kernel(float* __restrict__ p, int n) {
  int i = blockIdx.x * blockDim.x + threadIdx.x;
  if (i < n) p[i] = 0.0f;
}

// ---------------- edge kernel: radial MLP (WMMA) + CG contraction + scatter -
__global__ __launch_bounds__(256) void nequip_edge_kernel(
    const float* __restrict__ vectors, const int* __restrict__ senders,
    const int* __restrict__ receivers, const float* __restrict__ mfeat,
    const _Float16* __restrict__ bf0, const _Float16* __restrict__ bf1,
    const _Float16* __restrict__ bf2, float* __restrict__ agg) {
  __shared__ __align__(16) _Float16 hbuf_s[8][16*64]; // per-wave 16x64 activations
  __shared__ float ebuf_s[8][16*12];                  // per-wave edge metadata

  const int lane = threadIdx.x & 31;
  const int wib  = threadIdx.x >> 5;
  _Float16* hbuf = hbuf_s[wib];
  float*    ebuf = ebuf_s[wib];
  const int hi = lane >> 4;   // K-half / row-half select
  const int lo = lane & 15;   // row (A) / col (B,D)

  // Preload Wm0/Wm1 B fragments (loop-invariant)
  v16h bW0[4], bW1[4][2];
  #pragma unroll
  for (int t = 0; t < 4; ++t)
    bW0[t] = *(const v16h*)(bf0 + ((size_t)t*32 + lane)*16);
  #pragma unroll
  for (int t = 0; t < 4; ++t)
    #pragma unroll
    for (int c = 0; c < 2; ++c)
      bW1[t][c] = *(const v16h*)(bf1 + ((size_t)(t*2+c)*32 + lane)*16);

  v8f zero8;
  #pragma unroll
  for (int i = 0; i < 8; ++i) zero8[i] = 0.0f;

  const int waveId = blockIdx.x * 8 + wib;
  const int nWaves = gridDim.x * 8;
  const int nTiles = N_EDGES / 16;

  for (int tile = waveId; tile < nTiles; tile += nWaves) {
    // ---- per-edge geometry + radial basis (lanes duplicate by row) ----
    int e = tile*16 + lo;
    float vx = vectors[(size_t)e*3+0], vy = vectors[(size_t)e*3+1], vz = vectors[(size_t)e*3+2];
    float len = sqrtf(vx*vx + vy*vy + vz*vz);
    float safe = (len > 1e-9f) ? len : 1.0f;
    float rinv = __builtin_amdgcn_rcpf(safe);
    float ux = vx*rinv, uy = vy*rinv, uz = vz*rinv;
    float x  = fminf(len * 0.25f, 1.0f);
    float env = 0.5f * (__cosf(PI_F * x) + 1.0f);
    float pref = 0.70710678f * env * rinv;      // sqrt(2/R_MAX)*env/safe_r
    float rb[8];
    #pragma unroll
    for (int k = 0; k < 8; ++k) rb[k] = pref * __sinf(PI_F * x * (float)(k+1));

    if (lane < 16) {
      float* eb = ebuf + lo*12;
      eb[0] = 1.73205081f*uy; eb[1] = 1.73205081f*uz; eb[2] = 1.73205081f*ux;
      eb[3] = 3.87298335f*ux*uy; eb[4] = 3.87298335f*uy*uz;
      eb[5] = 1.11803399f*(3.0f*uz*uz - 1.0f);
      eb[6] = 3.87298335f*ux*uz; eb[7] = 1.93649167f*(ux*ux - uy*uy);
      eb[8]  = __int_as_float(senders[e]);
      eb[9]  = __int_as_float(receivers[e]);
      eb[10] = (len == 0.0f) ? 0.0f : 1.0f;
    }

    // ---- stage 1: rb(16x8 pad 32) @ Wm0 -> h(16x64) ----
    v16h a1;
    #pragma unroll
    for (int i = 0; i < 16; ++i) a1[i] = (_Float16)0.0f;
    if (lane < 16) {
      #pragma unroll
      for (int i = 0; i < 8; ++i) a1[i] = (_Float16)rb[i];
    }
    #pragma unroll
    for (int t = 0; t < 4; ++t) {
      v8f c = __builtin_amdgcn_wmma_f32_16x16x32_f16(false, a1, false, bW0[t],
                                                     (short)0, zero8, false, false);
      int col = t*16 + lo;
      #pragma unroll
      for (int r = 0; r < 8; ++r)
        hbuf[(hi*8 + r)*64 + col] = (_Float16)silu_f(c[r]);
    }
    __asm__ volatile("" ::: "memory");

    // ---- transpose D-layout -> A-layout via LDS ----
    v16h a2lo, a2hi;
    {
      int ko = hi*8;
      v8h p0 = *(const v8h*)(hbuf + lo*64 +      ko);
      v8h p1 = *(const v8h*)(hbuf + lo*64 + 16 + ko);
      v8h p2 = *(const v8h*)(hbuf + lo*64 + 32 + ko);
      v8h p3 = *(const v8h*)(hbuf + lo*64 + 48 + ko);
      #pragma unroll
      for (int i = 0; i < 8; ++i) { a2lo[i]=p0[i]; a2lo[8+i]=p1[i]; a2hi[i]=p2[i]; a2hi[8+i]=p3[i]; }
    }
    __asm__ volatile("" ::: "memory");

    // ---- stage 2: h @ Wm1 (K=64) ----
    #pragma unroll
    for (int t = 0; t < 4; ++t) {
      v8f c = __builtin_amdgcn_wmma_f32_16x16x32_f16(false, a2lo, false, bW1[t][0],
                                                     (short)0, zero8, false, false);
      c = __builtin_amdgcn_wmma_f32_16x16x32_f16(false, a2hi, false, bW1[t][1],
                                                 (short)0, c, false, false);
      int col = t*16 + lo;
      #pragma unroll
      for (int r = 0; r < 8; ++r)
        hbuf[(hi*8 + r)*64 + col] = (_Float16)silu_f(c[r]);
    }
    __asm__ volatile("" ::: "memory");

    v16h a3lo, a3hi;
    {
      int ko = hi*8;
      v8h p0 = *(const v8h*)(hbuf + lo*64 +      ko);
      v8h p1 = *(const v8h*)(hbuf + lo*64 + 16 + ko);
      v8h p2 = *(const v8h*)(hbuf + lo*64 + 32 + ko);
      v8h p3 = *(const v8h*)(hbuf + lo*64 + 48 + ko);
      #pragma unroll
      for (int i = 0; i < 8; ++i) { a3lo[i]=p0[i]; a3lo[8+i]=p1[i]; a3hi[i]=p2[i]; a3hi[8+i]=p3[i]; }
    }
    __asm__ volatile("" ::: "memory");

    // ---- stage 3: h @ Wm2 (K=64, 11 path tiles of 16 channels) ----
    v8f mix[11];
    #pragma unroll
    for (int p = 0; p < 11; ++p) {
      v16h blo = *(const v16h*)(bf2 + ((size_t)(p*2+0)*32 + lane)*16);
      v16h bhi = *(const v16h*)(bf2 + ((size_t)(p*2+1)*32 + lane)*16);
      v8f c = __builtin_amdgcn_wmma_f32_16x16x32_f16(false, a3lo, false, blo,
                                                     (short)0, zero8, false, false);
      mix[p] = __builtin_amdgcn_wmma_f32_16x16x32_f16(false, a3hi, false, bhi,
                                                      (short)0, c, false, false);
    }

    // ---- CG contraction + scatter (lane = channel, VGPR r = edge row) ----
    const int ch = lo;
    #pragma unroll
    for (int r = 0; r < 8; ++r) {
      const int mrow = hi*8 + r;
      const float* eb = ebuf + mrow*12;
      float Y[3][5];
      Y[0][0] = 1.0f;
      Y[1][0]=eb[0]; Y[1][1]=eb[1]; Y[1][2]=eb[2];
      Y[2][0]=eb[3]; Y[2][1]=eb[4]; Y[2][2]=eb[5]; Y[2][3]=eb[6]; Y[2][4]=eb[7];
      int snd = __float_as_int(eb[8]);
      int rcv = __float_as_int(eb[9]);
      float zf = eb[10];
      const float* mb = mfeat + (size_t)snd * FEAT;
      float X[3][5];
      X[0][0] = mb[ch];
      #pragma unroll
      for (int i = 0; i < 3; ++i) X[1][i] = mb[16 + ch*3 + i];
      #pragma unroll
      for (int i = 0; i < 5; ++i) X[2][i] = mb[64 + ch*5 + i];
      float O[3][5];
      #pragma unroll
      for (int a = 0; a < 3; ++a)
        #pragma unroll
        for (int b = 0; b < 5; ++b) O[a][b] = 0.0f;
      #pragma unroll
      for (int p = 0; p < 11; ++p) {
        const float w = mix[p][r] * zf;
        #pragma unroll
        for (int n = PSTART[p]; n < PSTART[p+1]; ++n)
          O[PL3[p]][CGK[n]] += w * X[PL1[p]][CGI[n]] * Y[PL2[p]][CGJ[n]]
                                 * (CGV[n] * PINVN[PL3[p]]);   // constant-folded
      }
      float* ab = agg + (size_t)rcv * FEAT;
      atomicAdd(&ab[ch], O[0][0]);
      #pragma unroll
      for (int i = 0; i < 3; ++i) atomicAdd(&ab[16 + ch*3 + i], O[1][i]);
      #pragma unroll
      for (int i = 0; i < 5; ++i) atomicAdd(&ab[64 + ch*5 + i], O[2][i]);
    }
  }
}

// ---------------- node down-projection + species skip + gating --------------
__global__ void nequip_down_kernel(const float* __restrict__ nf,
                                   const int* __restrict__ species,
                                   const float* __restrict__ agg,
                                   const float* __restrict__ Wd0,
                                   const float* __restrict__ Wd1,
                                   const float* __restrict__ Wd2,
                                   const float* __restrict__ Wsk0,
                                   const float* __restrict__ Wsk1,
                                   const float* __restrict__ Wsk2,
                                   float* __restrict__ out) {
  int idx = blockIdx.x * blockDim.x + threadIdx.x;
  if (idx >= N_NODES * 16) return;
  int n = idx >> 4, o = idx & 15;
  int sp = species[n];
  const float* f = nf + (size_t)n * FEAT;
  const float* a = agg + (size_t)n * FEAT;
  const float aggs = 0.25f; // 1/sqrt(AVG_NEIGH)
  const float inv  = 0.25f; // 1/sqrt(CH)
  float sv[3];
  #pragma unroll
  for (int g = 0; g < 3; ++g) {
    int col = g*16 + o;
    float acc = 0.f;
    const float* wk = Wsk0 + (size_t)sp * 16 * 48;
    #pragma unroll
    for (int c = 0; c < 16; ++c)
      acc += (a[c]*aggs) * Wd0[c*48 + col] + f[c] * wk[c*48 + col];
    sv[g] = acc * inv;
  }
  float sc = silu_f(sv[0]);
  float g1 = silu_f(sv[1]);
  float g2 = silu_f(sv[2]);
  float* oo = out + (size_t)n * FEAT;
  oo[o] = sc;
  const float* wk1 = Wsk1 + (size_t)sp * 256;
  #pragma unroll
  for (int i = 0; i < 3; ++i) {
    float acc = 0.f;
    #pragma unroll
    for (int c = 0; c < 16; ++c)
      acc += (a[16 + c*3 + i]*aggs) * Wd1[c*16 + o] + f[16 + c*3 + i] * wk1[c*16 + o];
    oo[16 + o*3 + i] = acc * inv * g1;
  }
  const float* wk2 = Wsk2 + (size_t)sp * 256;
  #pragma unroll
  for (int i = 0; i < 5; ++i) {
    float acc = 0.f;
    #pragma unroll
    for (int c = 0; c < 16; ++c)
      acc += (a[64 + c*5 + i]*aggs) * Wd2[c*16 + o] + f[64 + c*5 + i] * wk2[c*16 + o];
    oo[64 + o*5 + i] = acc * inv * g2;
  }
}

extern "C" void kernel_launch(void* const* d_in, const int* in_sizes, int n_in,
                              void* d_out, int out_size, void* d_ws, size_t ws_size,
                              hipStream_t stream) {
  const float* vectors    = (const float*)d_in[0];
  const float* node_feats = (const float*)d_in[1];
  const int*   species    = (const int*)d_in[2];
  const int*   senders    = (const int*)d_in[3];
  const int*   receivers  = (const int*)d_in[4];
  const float* Wu0 = (const float*)d_in[5];
  const float* Wu1 = (const float*)d_in[6];
  const float* Wu2 = (const float*)d_in[7];
  const float* Wm0 = (const float*)d_in[8];
  const float* Wm1 = (const float*)d_in[9];
  const float* Wm2 = (const float*)d_in[10];
  const float* Wd0 = (const float*)d_in[11];
  const float* Wd1 = (const float*)d_in[12];
  const float* Wd2 = (const float*)d_in[13];
  const float* Wsk0 = (const float*)d_in[14];
  const float* Wsk1 = (const float*)d_in[15];
  const float* Wsk2 = (const float*)d_in[16];
  float* out = (float*)d_out;

  const size_t M_BYTES = (size_t)N_NODES * FEAT * sizeof(float); // 18,874,368
  char* ws = (char*)d_ws;
  float* m_buf = (float*)ws;
  float* agg   = (float*)(ws + M_BYTES);
  _Float16* bf = (_Float16*)(ws + 2 * M_BYTES);
  _Float16* bf0 = bf;
  _Float16* bf1 = bf + 4 * 512;
  _Float16* bf2 = bf + 12 * 512;

  nequip_frag_kernel<<<(34*512 + 255)/256, 256, 0, stream>>>(Wm0, Wm1, Wm2, bf);
  nequip_up_kernel<<<(N_NODES*16 + 255)/256, 256, 0, stream>>>(node_feats, Wu0, Wu1, Wu2, m_buf);
  nequip_zero_kernel<<<(N_NODES*FEAT + 255)/256, 256, 0, stream>>>(agg, N_NODES*FEAT);
  nequip_edge_kernel<<<512, 256, 0, stream>>>(vectors, senders, receivers, m_buf,
                                              bf0, bf1, bf2, agg);
  nequip_down_kernel<<<(N_NODES*16 + 255)/256, 256, 0, stream>>>(
      node_feats, species, agg, Wd0, Wd1, Wd2, Wsk0, Wsk1, Wsk2, out);
}
